// SimplifiedRingAttention_76312978915794
// MI455X (gfx1250) — matrix operand order repaired and checked by reading, not compiled, not run on hardware
//
#include <hip/hip_runtime.h>

typedef __attribute__((ext_vector_type(16))) _Float16 v16h;
typedef __attribute__((ext_vector_type(8)))  _Float16 v8h;
typedef __attribute__((ext_vector_type(8)))  float    v8f;
typedef __attribute__((ext_vector_type(8)))  unsigned int v8u;

#define WMMA_F32_F16(A, B, C) \
  __builtin_amdgcn_wmma_f32_16x16x32_f16(false, (A), false, (B), (short)0, (C), false, false)

// v_cvt_pk_rtz_f16_f32: pack two f32 into one dword of f16 (a -> low, b -> high)
__device__ __forceinline__ unsigned int pkh(float a, float b) {
  return __builtin_bit_cast(unsigned int, __builtin_amdgcn_cvt_pkrtz(a, b));
}

// DPP row_ror:N rotate within the 16-lane DPP row (lanes 0-15 / 16-31 independent)
#define DPP_ROR(x, n) __builtin_bit_cast(float, \
    __builtin_amdgcn_update_dpp(0, __builtin_bit_cast(int, (x)), 0x120 | (n), 0xf, 0xf, true))

__device__ __forceinline__ float rowmax16(float x) {
  x = fmaxf(x, DPP_ROR(x, 8));
  x = fmaxf(x, DPP_ROR(x, 4));
  x = fmaxf(x, DPP_ROR(x, 2));
  x = fmaxf(x, DPP_ROR(x, 1));
  return x;
}

constexpr int BATCH = 2, SEQ = 2048, HEADS = 16, DHEAD = 64;
constexpr int ROWSTRIDE = HEADS * DHEAD;   // 1024 floats between tokens of one head
constexpr int PSTR = 72;                   // fast-path P LDS row stride (16x64 tile)
constexpr int PSTR_FB = 40;                // fallback P LDS row stride (16x32 tile)
constexpr int VSTR = 40;                   // fallback V^T LDS stride

// ============================================================================
// Prep kernel 1: f32 [b,n,h,d] -> f16 [b,h,n,d], optional scale (Q: 1/sqrt(d))
// ============================================================================
__global__ __launch_bounds__(256)
void cvt_permute_qk(const float* __restrict__ src, _Float16* __restrict__ dst, float scale) {
  const int t  = blockIdx.x * 256 + threadIdx.x;     // one float4 per thread
  const int d4 = t & 15;                             // D/4 = 16 groups
  const int n  = (t >> 4) & (SEQ - 1);
  const int bh = t >> 15;                            // 0..B*H-1
  const int b = bh >> 4, h = bh & 15;
  const float4 v = *(const float4*)(src + ((size_t)(b * SEQ + n) * HEADS + h) * DHEAD + d4 * 4);
  uint2 o;
  o.x = pkh(v.x * scale, v.y * scale);
  o.y = pkh(v.z * scale, v.w * scale);
  *(uint2*)(dst + ((size_t)bh * SEQ + n) * DHEAD + d4 * 4) = o;
}

// ============================================================================
// Prep kernel 2: V f32 [b,n,h,d] -> V^T f16 [b,h,d,n] via LDS-tiled transpose
// ============================================================================
__global__ __launch_bounds__(256)
void cvt_transpose_v(const float* __restrict__ src, _Float16* __restrict__ dst) {
  __shared__ alignas(16) _Float16 tile[64 * 72];
  const int t   = threadIdx.x;
  const int n0  = (blockIdx.x & 31) * 64;            // SEQ/64 = 32 tiles
  const int bh  = blockIdx.x >> 5;
  const int b = bh >> 4, h = bh & 15;

  const int nl = t >> 2;
  const int d0 = (t & 3) * 16;
  const float* srow = src + ((size_t)(b * SEQ + n0 + nl) * HEADS + h) * DHEAD + d0;
#pragma unroll
  for (int j = 0; j < 4; ++j) {
    const float4 v = *(const float4*)(srow + 4 * j);
    tile[(d0 + 4 * j + 0) * 72 + nl] = (_Float16)v.x;
    tile[(d0 + 4 * j + 1) * 72 + nl] = (_Float16)v.y;
    tile[(d0 + 4 * j + 2) * 72 + nl] = (_Float16)v.z;
    tile[(d0 + 4 * j + 3) * 72 + nl] = (_Float16)v.w;
  }
  __syncthreads();

  const int d  = t >> 2;
  const int nc = (t & 3) * 16;
  const v8h x0 = *(const v8h*)&tile[d * 72 + nc];
  const v8h x1 = *(const v8h*)&tile[d * 72 + nc + 8];
  _Float16* drow = dst + ((size_t)bh * DHEAD + d) * SEQ + n0 + nc;
  *(v8h*)drow       = x0;
  *(v8h*)(drow + 8) = x1;
}

// ============================================================================
// Fast attention, 64-key steps: 18 WMMAs/iter (8 QK + 8 PV + 2 row-sum),
// one softmax/rescale pass per 64 keys, f16 operands pre-staged in ws.
// ============================================================================
__global__ __launch_bounds__(32)
void fa_fwd_wmma_fast(const _Float16* __restrict__ Qh, const _Float16* __restrict__ Kh,
                      const _Float16* __restrict__ Vth, float* __restrict__ og,
                      const int* __restrict__ pcausal) {
  __shared__ alignas(16) _Float16 Pbuf[16 * PSTR];   // P tile, row-major [M][0..63]

  const int lane = threadIdx.x;
  const int hs   = lane >> 4;
  const int l15  = lane & 15;

  const int qt = blockIdx.x & (SEQ / 16 - 1);
  const int hd = (blockIdx.x >> 7) & (HEADS - 1);
  const int bi = blockIdx.x >> 11;
  const int q0 = qt * 16;

  const int causal = *pcausal;
  const int kmax   = causal ? (q0 + 16) : SEQ;

  const size_t bh = (size_t)bi * HEADS + hd;
  const _Float16* Qb = Qh + (bh * SEQ + q0) * DHEAD;
  const _Float16* Kb = Kh + bh * SEQ * DHEAD;
  const _Float16* Vb = Vth + bh * (size_t)DHEAD * SEQ;
  float* Ob = og + ((size_t)bi * SEQ * HEADS + hd) * DHEAD + (size_t)q0 * ROWSTRIDE;

  // ---- Q tile -> two 16x32 A tiles (already f16, pre-scaled) ----
  const int kb = hs * 16;
  v16h a0, a1;
  {
    const _Float16* qrow = Qb + (size_t)l15 * DHEAD;
    const v8h t0 = *(const v8h*)(qrow + kb);
    const v8h t1 = *(const v8h*)(qrow + kb + 8);
    const v8h t2 = *(const v8h*)(qrow + 32 + kb);
    const v8h t3 = *(const v8h*)(qrow + 40 + kb);
#pragma unroll
    for (int i = 0; i < 8; ++i) {
      a0[i] = t0[i]; a0[i + 8] = t1[i];
      a1[i] = t2[i]; a1[i + 8] = t3[i];
    }
  }

  v16h ones;
#pragma unroll
  for (int i = 0; i < 16; ++i) ones[i] = (_Float16)1.0f;

  float mrow[8], srow[8];
  v8f oacc[4] = {};
#pragma unroll
  for (int r = 0; r < 8; ++r) { mrow[r] = -1e30f; srow[r] = 0.f; }

  const int koff = hs * 16;
  for (int kb0 = 0; kb0 < kmax; kb0 += 64) {
    __syncthreads();  // Pbuf reads of previous iteration complete

    // ---- S = Q @ K^T : four 16-key score tiles, direct f16 B loads ----
    v8f st[4];
#pragma unroll
    for (int g = 0; g < 4; ++g) {
      const _Float16* krow = Kb + (size_t)(kb0 + g * 16 + l15) * DHEAD;
      const v16h b0 = *(const v16h*)(krow + koff);
      const v16h b1 = *(const v16h*)(krow + 32 + koff);
      v8f s = {};
      s = WMMA_F32_F16(a0, b0, s);
      s = WMMA_F32_F16(a1, b1, s);
      st[g] = s;
    }

    if (kb0 + 64 < kmax) {
      __builtin_prefetch(Kb + (size_t)(kb0 + 64 + lane) * DHEAD, 0, 1);
      __builtin_prefetch(Kb + (size_t)(kb0 + 96 + lane) * DHEAD, 0, 1);
      __builtin_prefetch(Vb + (size_t)lane * SEQ + kb0 + 64, 0, 1);
      __builtin_prefetch(Vb + (size_t)(lane + 32) * SEQ + kb0 + 64, 0, 1);
    }

    // causal mask only on diagonal-straddling blocks
    if (causal && (kb0 + 63 > q0)) {
#pragma unroll
      for (int g = 0; g < 4; ++g) {
        const int key = kb0 + g * 16 + l15;
#pragma unroll
        for (int r = 0; r < 8; ++r)
          if (key > q0 + (r + 8 * hs)) st[g][r] = -3.0e38f;
      }
    }

    // ---- online softmax: one pass per 64 keys ----
    float al[8];
#pragma unroll
    for (int r = 0; r < 8; ++r) {
      float t = fmaxf(fmaxf(st[0][r], st[1][r]), fmaxf(st[2][r], st[3][r]));
      t = rowmax16(t);
      const float mn = fmaxf(mrow[r], t);
      al[r]   = __expf(mrow[r] - mn);
      mrow[r] = mn;
      st[0][r] = __expf(st[0][r] - mn);
      st[1][r] = __expf(st[1][r] - mn);
      st[2][r] = __expf(st[2][r] - mn);
      st[3][r] = __expf(st[3][r] - mn);
    }
#pragma unroll
    for (int c = 0; c < 4; ++c)
#pragma unroll
      for (int r = 0; r < 8; ++r) oacc[c][r] *= al[r];

    // ---- P (C layout) -> LDS row-major [16][64] f16 ----
#pragma unroll
    for (int r = 0; r < 8; ++r) {
      const int M = r + 8 * hs;
      Pbuf[M * PSTR + l15]      = (_Float16)st[0][r];
      Pbuf[M * PSTR + 16 + l15] = (_Float16)st[1][r];
      Pbuf[M * PSTR + 32 + l15] = (_Float16)st[2][r];
      Pbuf[M * PSTR + 48 + l15] = (_Float16)st[3][r];
    }
    __syncthreads();

    // ---- P back as two 16x32 A tiles ----
    v16h pa0, pa1;
    {
      const v8h lo0 = *(const v8h*)&Pbuf[l15 * PSTR + kb];
      const v8h up0 = *(const v8h*)&Pbuf[l15 * PSTR + 8 + kb];
      const v8h lo1 = *(const v8h*)&Pbuf[l15 * PSTR + 32 + kb];
      const v8h up1 = *(const v8h*)&Pbuf[l15 * PSTR + 40 + kb];
#pragma unroll
      for (int i = 0; i < 8; ++i) {
        pa0[i] = lo0[i]; pa0[i + 8] = up0[i];
        pa1[i] = lo1[i]; pa1[i + 8] = up1[i];
      }
    }

    // ---- O += P @ V : B tiles straight from pre-transposed V^T ----
#pragma unroll
    for (int c = 0; c < 4; ++c) {
      const _Float16* vrow = Vb + (size_t)(c * 16 + l15) * SEQ + kb0 + koff;
      const v16h vb0 = *(const v16h*)vrow;          // keys kb0   .. kb0+31 slice
      const v16h vb1 = *(const v16h*)(vrow + 32);   // keys kb0+32.. kb0+63 slice
      oacc[c] = WMMA_F32_F16(pa0, vb0, oacc[c]);
      oacc[c] = WMMA_F32_F16(pa1, vb1, oacc[c]);
    }

    // ---- row sums on the matrix pipe ----
    v8f rsum = {};
    rsum = WMMA_F32_F16(pa0, ones, rsum);
    rsum = WMMA_F32_F16(pa1, ones, rsum);
#pragma unroll
    for (int r = 0; r < 8; ++r) srow[r] = srow[r] * al[r] + rsum[r];
  }

#pragma unroll
  for (int r = 0; r < 8; ++r) {
    const int M = r + 8 * hs;
    const float inv = 1.f / srow[r];
    float* orow = Ob + (size_t)M * ROWSTRIDE;
#pragma unroll
    for (int c = 0; c < 4; ++c) orow[c * 16 + l15] = oacc[c][r] * inv;
  }
}

// ============================================================================
// Fallback (self-contained, fp32 inputs) — used only if workspace is too small
// ============================================================================
__global__ __launch_bounds__(32)
void fa_fwd_wmma_fb(const float* __restrict__ qg, const float* __restrict__ kg,
                    const float* __restrict__ vg, float* __restrict__ og,
                    const int* __restrict__ pcausal) {
  __shared__ alignas(16) _Float16 Pbuf[16 * PSTR_FB];
  __shared__ alignas(16) _Float16 Vt[DHEAD * VSTR];

  const int lane = threadIdx.x;
  const int hs = lane >> 4, l15 = lane & 15;
  const int qt = blockIdx.x & (SEQ / 16 - 1);
  const int hd = (blockIdx.x >> 7) & (HEADS - 1);
  const int bi = blockIdx.x >> 11;
  const int q0 = qt * 16;
  const int causal = *pcausal;
  const int kmax = causal ? (q0 + 16) : SEQ;

  const size_t headoff = ((size_t)bi * SEQ * HEADS + hd) * DHEAD;
  const float* Qb = qg + headoff + (size_t)q0 * ROWSTRIDE;
  const float* Kb = kg + headoff;
  const float* Vb = vg + headoff;
  float* Ob = og + headoff + (size_t)q0 * ROWSTRIDE;

  const int kb = hs * 16;
  v16h a0, a1;
  {
    const float* qrow = Qb + (size_t)l15 * ROWSTRIDE;
    const float sc = 0.125f;
    v8u ua0, ua1;
#pragma unroll
    for (int j = 0; j < 4; ++j) {
      ua0[j]     = pkh(qrow[kb + 2*j] * sc,      qrow[kb + 2*j + 1] * sc);
      ua0[j + 4] = pkh(qrow[kb + 8 + 2*j] * sc, qrow[kb + 9 + 2*j] * sc);
      ua1[j]     = pkh(qrow[32 + kb + 2*j] * sc, qrow[32 + kb + 2*j + 1] * sc);
      ua1[j + 4] = pkh(qrow[40 + kb + 2*j] * sc, qrow[40 + kb + 2*j + 1] * sc);
    }
    a0 = __builtin_bit_cast(v16h, ua0);
    a1 = __builtin_bit_cast(v16h, ua1);
  }

  v16h ones;
#pragma unroll
  for (int i = 0; i < 16; ++i) ones[i] = (_Float16)1.0f;

  float mrow[8], srow[8];
  v8f oacc[4] = {};
#pragma unroll
  for (int r = 0; r < 8; ++r) { mrow[r] = -1e30f; srow[r] = 0.f; }

  for (int kb0 = 0; kb0 < kmax; kb0 += 32) {
    __syncthreads();
#pragma unroll
    for (int u = 0; u < 8; ++u) {
      const int kA = 4 * u + 2 * hs;
      const int d0 = l15 * 4;
      const float* ra = Vb + (size_t)(kb0 + kA) * ROWSTRIDE + d0;
      const float4 va  = *(const float4*)ra;
      const float4 vb4 = *(const float4*)(ra + ROWSTRIDE);
      *(unsigned int*)&Vt[(d0 + 0) * VSTR + kA] = pkh(va.x, vb4.x);
      *(unsigned int*)&Vt[(d0 + 1) * VSTR + kA] = pkh(va.y, vb4.y);
      *(unsigned int*)&Vt[(d0 + 2) * VSTR + kA] = pkh(va.z, vb4.z);
      *(unsigned int*)&Vt[(d0 + 3) * VSTR + kA] = pkh(va.w, vb4.w);
    }

    v8f st[2];
#pragma unroll
    for (int g = 0; g < 2; ++g) {
      const int koff = hs * 16;
      const float* krow = Kb + (size_t)(kb0 + g * 16 + l15) * ROWSTRIDE;
      v8u ub0, ub1;
#pragma unroll
      for (int j = 0; j < 8; ++j) {
        ub0[j] = pkh(krow[koff + 2*j],      krow[koff + 2*j + 1]);
        ub1[j] = pkh(krow[32 + koff + 2*j], krow[32 + koff + 2*j + 1]);
      }
      v8f s = {};
      s = WMMA_F32_F16(a0, __builtin_bit_cast(v16h, ub0), s);
      s = WMMA_F32_F16(a1, __builtin_bit_cast(v16h, ub1), s);
      st[g] = s;
    }

    if (causal && (kb0 + 31 > q0)) {
#pragma unroll
      for (int g = 0; g < 2; ++g) {
        const int key = kb0 + g * 16 + l15;
#pragma unroll
        for (int r = 0; r < 8; ++r)
          if (key > q0 + (r + 8 * hs)) st[g][r] = -3.0e38f;
      }
    }

    float al[8];
#pragma unroll
    for (int r = 0; r < 8; ++r) {
      const float t  = rowmax16(fmaxf(st[0][r], st[1][r]));
      const float mn = fmaxf(mrow[r], t);
      al[r] = __expf(mrow[r] - mn);
      mrow[r] = mn;
      st[0][r] = __expf(st[0][r] - mn);
      st[1][r] = __expf(st[1][r] - mn);
    }
#pragma unroll
    for (int c = 0; c < 4; ++c)
#pragma unroll
      for (int r = 0; r < 8; ++r) oacc[c][r] *= al[r];

#pragma unroll
    for (int r = 0; r < 8; ++r) {
      const int M = r + 8 * hs;
      Pbuf[M * PSTR_FB + l15]      = (_Float16)st[0][r];
      Pbuf[M * PSTR_FB + 16 + l15] = (_Float16)st[1][r];
    }
    __syncthreads();

    v16h pa;
    {
      const v8h lo = *(const v8h*)&Pbuf[l15 * PSTR_FB + kb];
      const v8h up = *(const v8h*)&Pbuf[l15 * PSTR_FB + 8 + kb];
#pragma unroll
      for (int i = 0; i < 8; ++i) { pa[i] = lo[i]; pa[i + 8] = up[i]; }
    }

    const int koff = hs * 16;
#pragma unroll
    for (int c = 0; c < 4; ++c) {
      const int dim = c * 16 + l15;
      const v8h x0 = *(const v8h*)&Vt[dim * VSTR + koff];
      const v8h x1 = *(const v8h*)&Vt[dim * VSTR + koff + 8];
      v16h vb;
#pragma unroll
      for (int i = 0; i < 8; ++i) { vb[i] = x0[i]; vb[i + 8] = x1[i]; }
      oacc[c] = WMMA_F32_F16(pa, vb, oacc[c]);
    }

    v8f rsum = {};
    rsum = WMMA_F32_F16(pa, ones, rsum);
#pragma unroll
    for (int r = 0; r < 8; ++r) srow[r] = srow[r] * al[r] + rsum[r];
  }

#pragma unroll
  for (int r = 0; r < 8; ++r) {
    const int M = r + 8 * hs;
    const float inv = 1.f / srow[r];
    float* orow = Ob + (size_t)M * ROWSTRIDE;
#pragma unroll
    for (int c = 0; c < 4; ++c) orow[c * 16 + l15] = oacc[c][r] * inv;
  }
}

extern "C" void kernel_launch(void* const* d_in, const int* in_sizes, int n_in,
                              void* d_out, int out_size, void* d_ws, size_t ws_size,
                              hipStream_t stream) {
  const float* q = (const float*)d_in[0];
  const float* k = (const float*)d_in[1];
  const float* v = (const float*)d_in[2];
  const int* is_causal = (const int*)d_in[3];
  float* out = (float*)d_out;

  const size_t per = (size_t)BATCH * HEADS * SEQ * DHEAD * sizeof(_Float16);  // 8.39 MB
  const int attn_blocks = BATCH * HEADS * (SEQ / 16);                         // 4096

  if (ws_size >= 3 * per) {
    _Float16* qh  = (_Float16*)d_ws;
    _Float16* kh  = (_Float16*)((char*)d_ws + per);
    _Float16* vth = (_Float16*)((char*)d_ws + 2 * per);
    const int cvt_blocks = (BATCH * HEADS * SEQ * DHEAD / 4) / 256;           // 4096
    cvt_permute_qk<<<cvt_blocks, 256, 0, stream>>>(q, qh, 0.125f);
    cvt_permute_qk<<<cvt_blocks, 256, 0, stream>>>(k, kh, 1.0f);
    cvt_transpose_v<<<BATCH * HEADS * (SEQ / 64), 256, 0, stream>>>(v, vth);
    fa_fwd_wmma_fast<<<attn_blocks, 32, 0, stream>>>(qh, kh, vth, out, is_causal);
  } else {
    fa_fwd_wmma_fb<<<attn_blocks, 32, 0, stream>>>(q, k, v, out, is_causal);
  }
}
